// RBFNN_19327352832510
// MI455X (gfx1250) — matrix operand order repaired and checked by reading, compiled, not run on hardware
//
#include <hip/hip_runtime.h>
#include <stdint.h>

// ---------------- types ----------------
typedef __attribute__((ext_vector_type(16))) __bf16    v16bf;
typedef __attribute__((ext_vector_type(16))) _Float16  v16h;
typedef __attribute__((ext_vector_type(8)))  float     v8f;
typedef __attribute__((ext_vector_type(4)))  float     f32x4;
typedef __attribute__((ext_vector_type(4)))  unsigned int u32x4;

union BFrag { u32x4 q[2]; v16bf v; };
union HFrag { u32x4 q[2]; v16h  v; };

#if __has_builtin(__builtin_amdgcn_exp2f)
#define EXP2F(t) __builtin_amdgcn_exp2f(t)
#else
#define EXP2F(t) exp2f(t)
#endif

#define N_ROWS 65536
#define DIMK   256
#define NCENT  1024
#define NOUT   64
#define CSTRIDE 528                    // 512B bf16 row data + 16B pad (bank-spread)
#define CENT_BYTES (2 * 2 * 32 * CSTRIDE)   // buf(2) x {hi,lo} x 32 rows = 67584
#define RBF_OFF    CENT_BYTES               // 16-byte aligned (67584 = 16*4224)
#define RBF_ROWLEN 40                       // f16 elems per row (80B, bank-spread)
#define SMEM_BYTES (CENT_BYTES + 8 * 16 * RBF_ROWLEN * 2)  // 77824

// ---------------- precompute kernels ----------------
__global__ void rbf_prep_centers(const float* __restrict__ c,
                                 __bf16* __restrict__ hi, __bf16* __restrict__ lo) {
  int i = blockIdx.x * 256 + threadIdx.x;          // C*D threads
  float f = c[i];
  __bf16 h = (__bf16)f;
  hi[i] = h;
  lo[i] = (__bf16)(f - (float)h);
}

__global__ void rbf_prep_w(const float* __restrict__ W, _Float16* __restrict__ Wh) {
  int i = blockIdx.x * 256 + threadIdx.x;          // O*C threads
  Wh[i] = (_Float16)W[i];
}

__global__ void rbf_prep_meta(const float* __restrict__ c, const float* __restrict__ sigma,
                              float* __restrict__ K1, float* __restrict__ K2) {
  int cc = blockIdx.x * 256 + threadIdx.x;         // 1024 threads
  float s = sigma[cc];
  float k1 = -1.44269504088896340736f / (2.0f * s * s + 1e-8f); // -log2(e)/(2s^2+eps)
  float cs = 0.0f;
  const float* row = c + (size_t)cc * DIMK;
  for (int d = 0; d < DIMK; ++d) cs = fmaf(row[d], row[d], cs);
  K1[cc] = k1;
  K2[cc] = cs * k1;
}

// ---------------- fused main kernel ----------------
__global__ __launch_bounds__(256) void rbf_main(
    const float*    __restrict__ x,
    const __bf16*   __restrict__ ch,
    const __bf16*   __restrict__ clo,
    const _Float16* __restrict__ Wh,
    const float*    __restrict__ K1,
    const float*    __restrict__ K2,
    const float*    __restrict__ bias,
    float*          __restrict__ out) {
  extern __shared__ __align__(16) unsigned char smem[];
  // [0, 67584): centers double buffer; [67584, 77824): per-wave rbf f16 tiles

  const int tid    = threadIdx.x;
  const int wave   = tid >> 5;
  const int lane   = tid & 31;
  const int lh     = lane & 15;
  const int khalf  = lane >> 4;
  const int m_base = blockIdx.x * 128 + wave * 16;

  _Float16* rbf_w = (_Float16*)(smem + RBF_OFF) + wave * 16 * RBF_ROWLEN;

  auto cent_ptr = [&](int buf, int arr, int r) -> unsigned char* {
    return smem + (size_t)(((buf * 2 + arr) * 32 + r) * CSTRIDE);
  };

  // issue one superchunk's async stage: 64 rows (32 hi + 32 lo), 1 row/instruction/wave
  auto issue_chunk = [&](int sc, int buf) {
#pragma unroll
    for (int j = 0; j < 8; ++j) {
      const int idx = wave * 8 + j;     // 0..63
      const int arr = idx >> 5;         // 0=hi, 1=lo
      const int r   = idx & 31;
      const unsigned char* src = (const unsigned char*)(arr ? clo : ch)
                                 + ((size_t)(sc * 32 + r) << 9); // row * 512B
      unsigned int ldsoff = (unsigned int)(uintptr_t)cent_ptr(buf, arr, r) + lane * 16u;
      unsigned int voff   = lane * 16u;
      asm volatile("global_load_async_to_lds_b128 %0, %1, %2"
                   :: "v"(ldsoff), "v"(voff), "s"(src) : "memory");
    }
  };

  // ---- load x A-fragments once (bf16 hi/lo split), accumulate x_sq ----
  v16bf a_hi[8], a_lo[8];
  float xsq = 0.0f;
  {
    const float* xrow = x + (size_t)(m_base + lh) * DIMK + khalf * 8;
#pragma unroll
    for (int kc = 0; kc < 8; ++kc) {
      const f32x4* xp = (const f32x4*)(xrow + kc * 32);
      f32x4 f0 = xp[0], f1 = xp[1], f2 = xp[4], f3 = xp[5]; // K {0..7} and {16..23}
      float fv[16];
#pragma unroll
      for (int i = 0; i < 4; ++i) { fv[i] = f0[i]; fv[4+i] = f1[i]; fv[8+i] = f2[i]; fv[12+i] = f3[i]; }
      v16bf h, l;
#pragma unroll
      for (int i = 0; i < 16; ++i) {
        __bf16 hb = (__bf16)fv[i];
        h[i] = hb;
        l[i] = (__bf16)(fv[i] - (float)hb);
        xsq  = fmaf(fv[i], fv[i], xsq);
      }
      a_hi[kc] = h; a_lo[kc] = l;
    }
  }
  const float xsq_full = xsq + __shfl_xor(xsq, 16, 32); // lane L: ||x_{row L&15}||^2

  v8f acc2[4];
#pragma unroll
  for (int ot = 0; ot < 4; ++ot) acc2[ot] = (v8f){0,0,0,0,0,0,0,0};

  // ---- software-pipelined main loop over 32-center superchunks ----
  issue_chunk(0, 0);
  for (int sc = 0; sc < 32; ++sc) {
    const int buf = sc & 1;
    asm volatile("s_wait_asynccnt 0" ::: "memory");  // my loads for buf done
    __syncthreads();  // all waves loaded buf; all waves done reading buf^1 (iter sc-1)
    if (sc + 1 < 32) issue_chunk(sc + 1, buf ^ 1);   // prefetch overlaps compute below

    const int c_base = sc * 32;

    // ---- GEMM1: two 16x16 cross tiles, K=256, bf16x3 split ----
#pragma unroll
    for (int ct = 0; ct < 2; ++ct) {
      v8f acc = (v8f){0,0,0,0,0,0,0,0};
      const unsigned char* brow_h = cent_ptr(buf, 0, ct * 16 + lh);
      const unsigned char* brow_l = cent_ptr(buf, 1, ct * 16 + lh);
#pragma unroll
      for (int kc = 0; kc < 8; ++kc) {
        const int off = kc * 64 + khalf * 16;  // B: K {0..7} & {16..23} per lane half
        BFrag bh, bl;
        bh.q[0] = *(const u32x4*)(brow_h + off);
        bh.q[1] = *(const u32x4*)(brow_h + off + 32);
        bl.q[0] = *(const u32x4*)(brow_l + off);
        bl.q[1] = *(const u32x4*)(brow_l + off + 32);
        acc = __builtin_amdgcn_wmma_f32_16x16x32_bf16(false, a_hi[kc], false, bh.v, (short)0, acc, false, false);
        acc = __builtin_amdgcn_wmma_f32_16x16x32_bf16(false, a_lo[kc], false, bh.v, (short)0, acc, false, false);
        acc = __builtin_amdgcn_wmma_f32_16x16x32_bf16(false, a_hi[kc], false, bl.v, (short)0, acc, false, false);
      }
      // epilogue: rbf = exp2(xsq*k1 + k2 - 2*k1*cross), store f16 tile to LDS
      const int cg = c_base + ct * 16 + lh;
      const float k1 = K1[cg], k2 = K2[cg];
      const float m2k1 = -2.0f * k1;
#pragma unroll
      for (int r = 0; r < 8; ++r) {
        const int row = r + ((lane & 16) >> 1);            // D-layout M index
        const float xr = __shfl(xsq_full, row, 32);
        const float t  = fmaf(acc[r], m2k1, fmaf(xr, k1, k2));
        rbf_w[row * RBF_ROWLEN + ct * 16 + lh] = (_Float16)EXP2F(t);
      }
    }

    // ---- GEMM2: acc2[16n x 64o] += rbf[16n x 32c] * W^T, f16 WMMA ----
    HFrag a2;
    {
      const _Float16* ap = rbf_w + lh * RBF_ROWLEN + khalf * 8; // A: K {0..7} & {16..23}
      a2.q[0] = *(const u32x4*)ap;
      a2.q[1] = *(const u32x4*)(ap + 16);
    }
#pragma unroll
    for (int ot = 0; ot < 4; ++ot) {
      const _Float16* wp = Wh + (size_t)(ot * 16 + lh) * NCENT + c_base + khalf * 8;
      HFrag b2;
      b2.q[0] = *(const u32x4*)wp;
      b2.q[1] = *(const u32x4*)(wp + 16);
      acc2[ot] = __builtin_amdgcn_wmma_f32_16x16x32_f16(false, a2.v, false, b2.v, (short)0, acc2[ot], false, false);
    }
  }

  // ---- write out + bias ----
#pragma unroll
  for (int ot = 0; ot < 4; ++ot) {
    const int o = ot * 16 + lh;
    const float bv = bias[o];
#pragma unroll
    for (int r = 0; r < 8; ++r) {
      const int row = m_base + r + ((lane & 16) >> 1);
      out[(size_t)row * NOUT + o] = acc2[ot][r] + bv;
    }
  }
}

// ---------------- launch ----------------
extern "C" void kernel_launch(void* const* d_in, const int* in_sizes, int n_in,
                              void* d_out, int out_size, void* d_ws, size_t ws_size,
                              hipStream_t stream) {
  const float* x       = (const float*)d_in[0];
  const float* centers = (const float*)d_in[1];
  const float* sigma   = (const float*)d_in[2];
  const float* W       = (const float*)d_in[3];
  const float* b       = (const float*)d_in[4];
  float* out = (float*)d_out;

  char* ws = (char*)d_ws;
  __bf16*    ch  = (__bf16*)ws;                              // 512 KB
  __bf16*    clo = (__bf16*)(ws + 512 * 1024);               // 512 KB
  _Float16*  Wh  = (_Float16*)(ws + 1024 * 1024);            // 128 KB
  float*     K1  = (float*)(ws + 1024 * 1024 + 128 * 1024);  // 4 KB
  float*     K2  = K1 + NCENT;                               // 4 KB

  rbf_prep_centers<<<(NCENT * DIMK) / 256, 256, 0, stream>>>(centers, ch, clo);
  rbf_prep_w<<<(NOUT * NCENT) / 256, 256, 0, stream>>>(W, Wh);
  rbf_prep_meta<<<NCENT / 256, 256, 0, stream>>>(centers, sigma, K1, K2);
  rbf_main<<<N_ROWS / 128, 256, SMEM_BYTES, stream>>>(x, ch, clo, Wh, K1, K2, b, out);
}
